// GridAttentionWeight_64364379898402
// MI455X (gfx1250) — compile-verified, matrix operand order checked by reading, not run on hardware
//
#include <hip/hip_runtime.h>
#include <hip/hip_bf16.h>
#include <stdint.h>

typedef __bf16 bf16_t;
typedef __bf16 v2bf  __attribute__((ext_vector_type(2)));
typedef __bf16 v16bf __attribute__((ext_vector_type(16)));
typedef float  v8f   __attribute__((ext_vector_type(8)));

#define NPOS   16384   // T*H*W
#define HWPIX  4096    // H*W
#define SLOTS  130
#define LDSTR  72      // LDS row stride (bf16): 144B rows -> 16B aligned + conflict-free

// ---- WMMA operand load ------------------------------------------------------
// Both A ([m][k]) and B ([n][k]) tiles stored row-major, K contiguous.
// ISA 7.12.2 K mapping: element j of lane L (half=L>>4) holds
//   K = k0 + 8*half + j            (j<8)   -> 16B contiguous
//   K = k0 + 16 + 8*half + (j-8)   (j>=8)  -> 16B contiguous at +16 elems
__device__ __forceinline__ v16bf load_mk(const bf16_t* lds, int rowBase, int k0) {
  const int lane = threadIdx.x & 31;
  const int half = lane >> 4;
  const bf16_t* p = lds + (rowBase + (lane & 15)) * LDSTR + k0 + 8 * half;
  v16bf v;
  *(uint4*)&v       = *(const uint4*)(p);        // ds_load_b128
  *((uint4*)&v + 1) = *(const uint4*)(p + 16);   // ds_load_b128
  return v;
}

#define WMMA_BF16(A, B, C) \
  __builtin_amdgcn_wmma_f32_16x16x32_bf16(false, (A), false, (B), (short)0, (C), false, false)

// ---- CDNA5 async global->LDS copy (ASYNCcnt) --------------------------------
__device__ __forceinline__ void async_b128(uint32_t lds_off, const void* gaddr) {
  asm volatile("global_load_async_to_lds_b128 %0, %1, off"
               :: "v"(lds_off), "v"((uint64_t)(uintptr_t)gaddr) : "memory");
}
__device__ __forceinline__ void wait_async0() {
  asm volatile("s_wait_asynccnt 0" ::: "memory");
}

// ---- Kernel 1: projection GEMM ----------------------------------------------
// Yh layout: [head][pos][p64] bf16, idx = head*1048576 + pos*64 + p
// grid (NPOS/64, 4 o-tiles), 256 threads (8 waves)
__global__ __launch_bounds__(256) void proj_kernel(const float* __restrict__ Wm,
                                                   const float* __restrict__ bias,
                                                   const float* __restrict__ X,
                                                   bf16_t* __restrict__ Yh) {
  __shared__ bf16_t As[64 * LDSTR];   // [m][k]
  __shared__ bf16_t BsT[64 * LDSTR];  // [n][k]
  const int tid = threadIdx.x;
  const int wave = tid >> 5;
  const int lane = tid & 31;
  const int posBase = blockIdx.x * 64;
  const int oBase = blockIdx.y * 64;

  const int t0 = wave * 2, t1 = wave * 2 + 1;
  const int m0 = (t0 >> 2) * 16, n0 = (t0 & 3) * 16;
  const int m1 = (t1 >> 2) * 16, n1 = (t1 & 3) * 16;
  v8f acc0 = {}; v8f acc1 = {};

  for (int kc = 0; kc < 4; ++kc) {
    __syncthreads();
    for (int e = tid; e < 2048; e += 256) {        // A: [m][k] pairs (contiguous src)
      int m = e >> 5, k2 = (e & 31) * 2;
      const float* wp = Wm + (oBase + m) * 256 + kc * 64 + k2;
      v2bf h; h[0] = (bf16_t)wp[0]; h[1] = (bf16_t)wp[1];
      *(v2bf*)&As[m * LDSTR + k2] = h;
    }
    for (int e = tid; e < 2048; e += 256) {        // B^T: [n][k] pairs (coalesced in n)
      int n = e & 63, k2 = (e >> 6) * 2;
      float x0 = X[(kc * 64 + k2) * NPOS + posBase + n];
      float x1 = X[(kc * 64 + k2 + 1) * NPOS + posBase + n];
      v2bf h; h[0] = (bf16_t)x0; h[1] = (bf16_t)x1;
      *(v2bf*)&BsT[n * LDSTR + k2] = h;
    }
    __syncthreads();
    {
      v16bf a, b;
      a = load_mk(As, m0, 0);  b = load_mk(BsT, n0, 0);  acc0 = WMMA_BF16(a, b, acc0);
      a = load_mk(As, m0, 32); b = load_mk(BsT, n0, 32); acc0 = WMMA_BF16(a, b, acc0);
      a = load_mk(As, m1, 0);  b = load_mk(BsT, n1, 0);  acc1 = WMMA_BF16(a, b, acc1);
      a = load_mk(As, m1, 32); b = load_mk(BsT, n1, 32); acc1 = WMMA_BF16(a, b, acc1);
    }
  }

  // packed bf16 store: lane holds 8 consecutive p-channels -> one 16B store
  const int nIdx = lane & 15, half = lane >> 4;
  const int head = oBase >> 6;
  {
    union { uint4 u; bf16_t h[8]; } r;
#pragma unroll
    for (int jj = 0; jj < 8; ++jj) r.h[jj] = (bf16_t)(acc0[jj] + bias[oBase + m0 + jj + 8 * half]);
    *(uint4*)(Yh + head * 1048576 + (posBase + n0 + nIdx) * 64 + m0 + 8 * half) = r.u;
  }
  {
    union { uint4 u; bf16_t h[8]; } r;
#pragma unroll
    for (int jj = 0; jj < 8; ++jj) r.h[jj] = (bf16_t)(acc1[jj] + bias[oBase + m1 + jj + 8 * half]);
    *(uint4*)(Yh + head * 1048576 + (posBase + n1 + nIdx) * 64 + m1 + 8 * half) = r.u;
  }
}

// ---- Kernel 1b: pre-convert rel-pos embeddings to padded bf16 ---------------
__global__ __launch_bounds__(256) void emb_prep(const float* __restrict__ ye,
                                                const float* __restrict__ xe,
                                                bf16_t* __restrict__ yh,
                                                bf16_t* __restrict__ xh) {
  int i = blockIdx.x * 256 + threadIdx.x;   // 8192 = 128 rows x 64
  int r = i >> 6;
  yh[i] = (r < 127) ? (bf16_t)ye[i] : (bf16_t)0.0f;
  xh[i] = (r < 127) ? (bf16_t)xe[i] : (bf16_t)0.0f;
}

// ---- Kernel 2: temporal energy (slots 0..3) ---------------------------------
__global__ __launch_bounds__(256) void et_kernel(const bf16_t* __restrict__ pqh,
                                                 const bf16_t* __restrict__ pkh,
                                                 const float* __restrict__ temb,
                                                 const int* __restrict__ frm,
                                                 float* __restrict__ out) {
  int gid = blockIdx.x * 256 + threadIdx.x;
  int b = gid >> 12;
  int pix = gid & 4095;
  int f[4];
#pragma unroll
  for (int i = 0; i < 4; ++i) f[i] = frm[i];

  const int base = b * 1048576 + pix * 64;     // [head][pos][p]
  float aQK[4][4] = {};   // [s][t]
  float aQE[4][4] = {};
  for (int c = 0; c < 8; ++c) {                // p in chunks of 8 (16B vector loads)
    union { uint4 v[4]; bf16_t h[32]; } qu, ku;
#pragma unroll
    for (int t = 0; t < 4; ++t) {
      qu.v[t] = *(const uint4*)(pqh + base + t * 262144 + c * 8);
      ku.v[t] = *(const uint4*)(pkh + base + t * 262144 + c * 8);
    }
#pragma unroll
    for (int pp = 0; pp < 8; ++pp) {
      float qv[4], kv[4];
#pragma unroll
      for (int t = 0; t < 4; ++t) { qv[t] = (float)qu.h[t * 8 + pp]; kv[t] = (float)ku.h[t * 8 + pp]; }
      int p = c * 8 + pp;
#pragma unroll
      for (int s = 0; s < 4; ++s)
#pragma unroll
        for (int t = 0; t < 4; ++t) {
          aQK[s][t] += qv[t] * kv[s];
          aQE[s][t] += qv[t] * temb[(f[s] - f[t] + 15) * 64 + p];
        }
    }
  }
#pragma unroll
  for (int s = 0; s < 4; ++s)
#pragma unroll
    for (int t = 0; t < 4; ++t)
      out[(size_t)((b * SLOTS + s) * 4 + t) * HWPIX + pix] = aQK[s][t] + aQE[s][t];
}

// ---- Kernel 3: row/col energy (slots 4..66 y, 67..129 x) --------------------
// Gram[g,n] = sum_p KsT[g][p]*QsT[n][p]^T-op ; D2[r,n] = sum_p Es[r][p]*q
// out[i,n] = Gram[g,n] + D2[g-n+63,n], g = i + (i>=n)
__global__ __launch_bounds__(256) void eyx_kernel(const bf16_t* __restrict__ pqh,
                                                  const bf16_t* __restrict__ pkh,
                                                  const bf16_t* __restrict__ embh,
                                                  float* __restrict__ out,
                                                  int slotBase, int isY) {
  __shared__ bf16_t QsT[64 * LDSTR];   // [i][p] : B operand rows
  __shared__ bf16_t KsT[64 * LDSTR];   // [i][p] : A operand rows (gram)
  __shared__ bf16_t Es[128 * LDSTR];   // [r][p] : A operand rows (emb)
  __shared__ float  GD[192 * 16];      // per n-tile: 0..63 gram(g), 64..191 emb(r)

  const int tid = threadIdx.x;
  const int wave = tid >> 5;
  const int sid = blockIdx.x;
  const int b = sid >> 8;
  const int t = (sid >> 6) & 3;
  const int j = sid & 63;                       // w for e_y, h for e_x
  const int iStride = isY ? 64 : 1;             // output stride along the free axis
  const int gStride = isY ? 4096 : 64;          // bf16-element stride between i-rows
  const int sliceBase = (b * 16384 + t * 4096 + (isY ? j : j * 64)) * 64;
  const int outBase = ((b * SLOTS + slotBase) * 4 + t) * HWPIX + (isY ? j : j * 64);

  const uint32_t qofs = (uint32_t)(uintptr_t)(&QsT[0]);
  const uint32_t kofs = (uint32_t)(uintptr_t)(&KsT[0]);
  const uint32_t eofs = (uint32_t)(uintptr_t)(&Es[0]);

  // async fill: every row is 16B-contiguous in global (p fastest in pqh layout)
  for (int e = tid; e < 512; e += 256) {
    int i = e >> 3, p8 = (e & 7) * 8;
    uint32_t lofs = (uint32_t)(i * LDSTR + p8) * 2;
    async_b128(qofs + lofs, pqh + sliceBase + i * gStride + p8);
    async_b128(kofs + lofs, pkh + sliceBase + i * gStride + p8);
  }
  for (int e = tid; e < 1024; e += 256) {
    int r = e >> 3, p8 = (e & 7) * 8;
    async_b128(eofs + (uint32_t)(r * LDSTR + p8) * 2, embh + r * 64 + p8);
  }
  wait_async0();

  for (int nT = 0; nT < 4; ++nT) {
    __syncthreads();                       // joins async fill / protects GD reuse
    const int nBase = nT * 16;
    v16bf b0 = load_mk(QsT, nBase, 0);     // shared B for all 12 A-tiles
    v16bf b1 = load_mk(QsT, nBase, 32);
    for (int tile = wave; tile < 12; tile += 8) {     // wave-uniform
      const bf16_t* Am = (tile < 4) ? KsT : Es;
      const int mBase = (tile < 4) ? tile * 16 : (tile - 4) * 16;
      v8f acc = {};
      v16bf a0 = load_mk(Am, mBase, 0);
      acc = WMMA_BF16(a0, b0, acc);
      v16bf a1 = load_mk(Am, mBase, 32);
      acc = WMMA_BF16(a1, b1, acc);
      const int lane = tid & 31;
      const int n = lane & 15, half = lane >> 4;
      const int rowBase = (tile < 4) ? tile * 16 : 64 + (tile - 4) * 16;
#pragma unroll
      for (int e2 = 0; e2 < 8; ++e2)
        GD[(rowBase + e2 + 8 * half) * 16 + n] = acc[e2];
    }
    __syncthreads();
    for (int e = tid; e < 63 * 16; e += 256) {   // combine + diagonal exclusion
      int i = e >> 4;                 // 0..62 output slot offset
      int nn = e & 15;
      int h = nBase + nn;             // self index along axis
      int g = i + ((i >= h) ? 1 : 0);
      int r = g - h + 63;             // 0..126
      out[outBase + i * NPOS + h * iStride] = GD[g * 16 + nn] + GD[(64 + r) * 16 + nn];
    }
  }
}

// ---- Kernel 4: in-place softmax over 130 slots ------------------------------
__global__ __launch_bounds__(256) void softmax_kernel(float* __restrict__ out) {
  int gid = blockIdx.x * 256 + threadIdx.x;
  int b = gid >> 14;
  int rem = gid & 16383;
  float* col = out + (size_t)b * SLOTS * NPOS + rem;
  float m = -3.4e38f;
  for (int s = 0; s < SLOTS; ++s) m = fmaxf(m, col[s * NPOS]);
  float sum = 0.f;
  for (int s = 0; s < SLOTS; ++s) sum += __expf(col[s * NPOS] - m);
  float inv = 1.f / sum;
  for (int s = 0; s < SLOTS; ++s) col[s * NPOS] = __expf(col[s * NPOS] - m) * inv;
}

// ---- Host launcher ----------------------------------------------------------
extern "C" void kernel_launch(void* const* d_in, const int* in_sizes, int n_in,
                              void* d_out, int out_size, void* d_ws, size_t ws_size,
                              hipStream_t stream) {
  (void)in_sizes; (void)n_in; (void)out_size; (void)ws_size;
  const float* query    = (const float*)d_in[0];
  const float* key_feat = (const float*)d_in[1];
  const int*   frm      = (const int*)d_in[2];
  const float* qw       = (const float*)d_in[3];
  const float* qb       = (const float*)d_in[4];
  const float* kw       = (const float*)d_in[5];
  const float* kb       = (const float*)d_in[6];
  const float* x_emb    = (const float*)d_in[7];
  const float* y_emb    = (const float*)d_in[8];
  const float* t_emb    = (const float*)d_in[9];
  float* out = (float*)d_out;

  bf16_t* pqh = (bf16_t*)d_ws;          // [4][16384][64] bf16 = 8 MiB
  bf16_t* pkh = pqh + 4 * 1048576;      // 8 MiB
  bf16_t* eyh = pkh + 4 * 1048576;      // 128*64 bf16
  bf16_t* exh = eyh + 8192;

  dim3 pgrid(NPOS / 64, 4);
  proj_kernel<<<pgrid, 256, 0, stream>>>(qw, qb, query, pqh);
  proj_kernel<<<pgrid, 256, 0, stream>>>(kw, kb, key_feat, pkh);
  emb_prep<<<32, 256, 0, stream>>>(y_emb, x_emb, eyh, exh);
  et_kernel<<<64, 256, 0, stream>>>(pqh, pkh, t_emb, frm, out);
  eyx_kernel<<<1024, 256, 0, stream>>>(pqh, pkh, eyh, out, 4, 1);   // e_y
  eyx_kernel<<<1024, 256, 0, stream>>>(pqh, pkh, exh, out, 67, 0);  // e_x
  softmax_kernel<<<256, 256, 0, stream>>>(out);
}
